// ImageLinearAttention_79525614452956
// MI455X (gfx1250) — matrix operand order repaired and checked by reading, not compiled, not run on hardware
//
#include <hip/hip_runtime.h>

// ---------------------------------------------------------------------------
// Image linear attention on MI455X (gfx1250), bf16 WMMA pipeline.
//
// Stages:
//   k1: Q/K/V = (W*x + b) [*scale]  -> bf16 workspace   (GEMM, wmma bf16)
//   k2: K <- softmax over n (row-wise, in place)        (reduction kernel)
//   k3: ctx[b,h] = softmax(K) @ V^T (64x64)             (wmma + split-K atomics)
//   k4: y = Wo @ concat_h( ctx_h^T @ softmax_d(Q_h) ) + bo   (fused in LDS)
// ---------------------------------------------------------------------------

typedef __attribute__((ext_vector_type(16))) __bf16 bf16x16;
typedef __attribute__((ext_vector_type(8)))  __bf16 bf16x8;
typedef __attribute__((ext_vector_type(4)))  __bf16 bf16x4;
typedef __attribute__((ext_vector_type(8)))  float  f32x8;

#define NB   4
#define CIN  128
#define NTOK 16384        // H*W
#define NHEAD 8
#define DK   64
#define PROJROWS 512      // NHEAD*DK
#define SCALE_QK 0.35355339059327373f   // 64^-0.25

__device__ __forceinline__ f32x8 zero8() {
  f32x8 z;
#pragma unroll
  for (int i = 0; i < 8; ++i) z[i] = 0.0f;
  return z;
}

__device__ __forceinline__ bf16x16 cat8(bf16x8 lo, bf16x8 hi) {
  return __builtin_shufflevector(lo, hi, 0,1,2,3,4,5,6,7,8,9,10,11,12,13,14,15);
}

// A matrix fragment, 16x32 bf16, source stored [M][K] (K contiguous, stride ld).
// Lane L: m = L%16, half = L/16; elem i -> K = half*16 + i (one 32B contiguous run).
__device__ __forceinline__ bf16x16 load_frag_a(const __bf16* base, int ld) {
  int lane = threadIdx.x & 31;
  int m = lane & 15, half = lane >> 4;
  const __bf16* r = base + m * ld;
  bf16x8 lo = *(const bf16x8*)(r + half * 16);
  bf16x8 hi = *(const bf16x8*)(r + half * 16 + 8);
  return cat8(lo, hi);
}

// B matrix fragment, 32x16 bf16, source stored [N][K] (K contiguous, stride ld).
// Lane L: n = L%16, half = L/16; elem i -> K = half*16 + i (one 32B contiguous run).
__device__ __forceinline__ bf16x16 load_frag_b(const __bf16* base, int ld) {
  int lane = threadIdx.x & 31;
  int n = lane & 15, half = lane >> 4;
  const __bf16* r = base + n * ld + half * 16;
  bf16x8 lo = *(const bf16x8*)(r);
  bf16x8 hi = *(const bf16x8*)(r + 8);
  return cat8(lo, hi);
}

__device__ __forceinline__ f32x8 wmma_bf16(bf16x16 a, bf16x16 b, f32x8 c) {
  return __builtin_amdgcn_wmma_f32_16x16x32_bf16(false, a, false, b, (short)0, c,
                                                 false, false);
}

// ---------------------------------------------------------------------------
// Kernel 1: QKV projections. Out[r,p] = (sum_c W[r,c]*x[b,c,p] + bias[r])*scale
// Block tile: 64 rows x 128 cols, kc = 32. 256 threads = 8 waves,
// wave w owns column tile w and all four 16-row tiles (4 accumulators).
// ---------------------------------------------------------------------------
__global__ __launch_bounds__(256) void qkv_proj_kernel(
    const float* __restrict__ x,
    const float* __restrict__ Wq, const float* __restrict__ bq,
    const float* __restrict__ Wk, const float* __restrict__ bk,
    const float* __restrict__ Wv, const float* __restrict__ bv,
    __bf16* __restrict__ Qb, __bf16* __restrict__ Kb, __bf16* __restrict__ Vb)
{
  __shared__ __bf16 As[64 * 32];    // [row][kc]
  __shared__ __bf16 Bs[128 * 32];   // [p][kc]  (transposed stage of x)

  int id = blockIdx.x;
  int colblk = id & 127; id >>= 7;
  int rowblk = id & 7;   id >>= 3;
  int proj = id % 3;
  int b = id / 3;

  const float* W; const float* bias; __bf16* Out; float scale;
  if (proj == 0)      { W = Wq; bias = bq; Out = Qb; scale = SCALE_QK; }
  else if (proj == 1) { W = Wk; bias = bk; Out = Kb; scale = SCALE_QK; }
  else                { W = Wv; bias = bv; Out = Vb; scale = 1.0f; }

  int r0 = rowblk * 64, p0 = colblk * 128;
  int tid = threadIdx.x, lane = tid & 31, wave = tid >> 5;
  const float* xb = x + (size_t)b * CIN * NTOK;

  f32x8 acc[4];
#pragma unroll
  for (int i = 0; i < 4; ++i) acc[i] = zero8();

  for (int c0 = 0; c0 < CIN; c0 += 32) {
#pragma unroll
    for (int i = 0; i < 8; ++i) {               // stage A: 64x32
      int idx = tid + 256 * i;
      int r = idx >> 5, kk = idx & 31;
      As[r * 32 + kk] = (__bf16)W[(r0 + r) * CIN + c0 + kk];
    }
#pragma unroll
    for (int i = 0; i < 16; ++i) {              // stage B: 128x32 (transpose)
      int idx = tid + 256 * i;
      int kk = idx >> 7, p = idx & 127;
      Bs[p * 32 + kk] = (__bf16)xb[(size_t)(c0 + kk) * NTOK + p0 + p];
    }
    __syncthreads();

    bf16x16 bfrag = load_frag_b(&Bs[wave * 16 * 32], 32);
#pragma unroll
    for (int rt = 0; rt < 4; ++rt) {
      bf16x16 afrag = load_frag_a(&As[rt * 16 * 32], 32);
      acc[rt] = wmma_bf16(afrag, bfrag, acc[rt]);
    }
    __syncthreads();
  }

  int half = lane >> 4, nn = lane & 15;
  int p = p0 + wave * 16 + nn;
#pragma unroll
  for (int rt = 0; rt < 4; ++rt) {
#pragma unroll
    for (int r = 0; r < 8; ++r) {
      int m = r0 + rt * 16 + r + 8 * half;
      float v = (acc[rt][r] + bias[m]) * scale;
      Out[(size_t)(b * PROJROWS + m) * NTOK + p] = (__bf16)v;
    }
  }
}

// ---------------------------------------------------------------------------
// Kernel 2: row softmax of K over n (16384), in place, bf16. One block per row.
// ---------------------------------------------------------------------------
__global__ __launch_bounds__(256) void ksoftmax_kernel(__bf16* __restrict__ K)
{
  __shared__ float red[256];
  int row = blockIdx.x;
  __bf16* base = K + (size_t)row * NTOK;
  int t = threadIdx.x;

  float mx = -3.4e38f;
#pragma unroll
  for (int c = 0; c < 8; ++c) {
    bf16x8 v = *(const bf16x8*)(base + (c * 256 + t) * 8);
#pragma unroll
    for (int j = 0; j < 8; ++j) mx = fmaxf(mx, (float)v[j]);
  }
  red[t] = mx; __syncthreads();
  for (int s = 128; s > 0; s >>= 1) {
    if (t < s) red[t] = fmaxf(red[t], red[t + s]);
    __syncthreads();
  }
  float rowmax = red[0]; __syncthreads();

  float sum = 0.0f;
#pragma unroll
  for (int c = 0; c < 8; ++c) {
    bf16x8 v = *(const bf16x8*)(base + (c * 256 + t) * 8);
#pragma unroll
    for (int j = 0; j < 8; ++j) sum += __expf((float)v[j] - rowmax);
  }
  red[t] = sum; __syncthreads();
  for (int s = 128; s > 0; s >>= 1) {
    if (t < s) red[t] += red[t + s];
    __syncthreads();
  }
  float inv = 1.0f / red[0]; __syncthreads();

#pragma unroll
  for (int c = 0; c < 8; ++c) {
    bf16x8 v = *(const bf16x8*)(base + (c * 256 + t) * 8);
    bf16x8 o;
#pragma unroll
    for (int j = 0; j < 8; ++j)
      o[j] = (__bf16)(__expf((float)v[j] - rowmax) * inv);
    *(bf16x8*)(base + (c * 256 + t) * 8) = o;
  }
}

__global__ void zero_f32_kernel(float* __restrict__ p, int n)
{
  int i = blockIdx.x * blockDim.x + threadIdx.x;
  if (i < n) p[i] = 0.0f;
}

// ---------------------------------------------------------------------------
// Kernel 3: ctx[b,h,d,e] = sum_n Ks[b,h,d,n] * V[b,h,e,n].
// Split-K over n (64 partitions of 256), f32 atomics into ctx.
// 256 threads = 8 waves; wave w: d-tile = w&3, two e-tiles at (w>>2)*32.
// A/B fragments load straight from global (both stored K-contiguous).
// ---------------------------------------------------------------------------
__global__ __launch_bounds__(256) void ctx_kernel(
    const __bf16* __restrict__ Ks, const __bf16* __restrict__ Vb,
    float* __restrict__ ctx)
{
  int bh = blockIdx.x >> 6;      // 0..31
  int split = blockIdx.x & 63;   // 0..63
  int tid = threadIdx.x, lane = tid & 31, wave = tid >> 5;
  int d_t = wave & 3;
  int e0 = (wave >> 2) * 32;

  const __bf16* aBase  = Ks + (size_t)(bh * 64 + d_t * 16) * NTOK;
  const __bf16* bBase0 = Vb + (size_t)(bh * 64 + e0) * NTOK;
  const __bf16* bBase1 = Vb + (size_t)(bh * 64 + e0 + 16) * NTOK;

  f32x8 acc0 = zero8(), acc1 = zero8();
  int n0 = split * 256;
#pragma unroll
  for (int s = 0; s < 8; ++s) {
    int n = n0 + s * 32;
    bf16x16 a  = load_frag_a(aBase + n, NTOK);
    bf16x16 b0 = load_frag_b(bBase0 + n, NTOK);
    bf16x16 b1 = load_frag_b(bBase1 + n, NTOK);
    acc0 = wmma_bf16(a, b0, acc0);
    acc1 = wmma_bf16(a, b1, acc1);
  }

  float* cb = ctx + (size_t)bh * 4096;
  int half = lane >> 4, nn = lane & 15;
#pragma unroll
  for (int r = 0; r < 8; ++r) {
    int d = d_t * 16 + r + 8 * half;
    atomicAdd(&cb[d * 64 + e0 + nn], acc0[r]);
    atomicAdd(&cb[d * 64 + e0 + 16 + nn], acc1[r]);
  }
}

// ---------------------------------------------------------------------------
// Kernel 4: per 64-column block of tokens:
//   qs = softmax_d(Q) (per head, in LDS);  out_h = ctx_h^T @ qs_h;
//   y = Wo @ concat_h(out_h) + bo.   All intermediates live in LDS (144 KB).
// ---------------------------------------------------------------------------
__global__ __launch_bounds__(256) void attn_out_kernel(
    const __bf16* __restrict__ Qb, const float* __restrict__ ctx,
    const float* __restrict__ Wo, const float* __restrict__ bo,
    float* __restrict__ y)
{
  __shared__ __bf16 qs[64 * 512];     // [p][r=h*64+d]   64 KB
  __shared__ __bf16 outf[64 * 512];   // [p][g=h*64+e]   64 KB
  __shared__ __bf16 ctxT[64 * 64];    // [e][d]           8 KB
  __shared__ __bf16 woA[128 * 32];    // [c][kc]          8 KB

  int b = blockIdx.x >> 8;
  int pblk = blockIdx.x & 255;
  int p0 = pblk * 64;
  int tid = threadIdx.x, lane = tid & 31, wave = tid >> 5;

  // Phase A: stage Q (transposed) into LDS.
  const __bf16* Qbase = Qb + (size_t)b * PROJROWS * NTOK + p0;
#pragma unroll
  for (int i = 0; i < 32; ++i) {
    int idx = tid + 256 * i;          // 8192 tasks of 4 tokens
    int r = idx >> 4;                 // 0..511
    int pc = idx & 15;                // 4-token chunk
    bf16x4 v = *(const bf16x4*)(Qbase + (size_t)r * NTOK + pc * 4);
#pragma unroll
    for (int j = 0; j < 4; ++j) qs[(pc * 4 + j) * 512 + r] = v[j];
  }
  __syncthreads();

  // Phase A2: softmax over d (64) per (head, token).
#pragma unroll
  for (int k = 0; k < 2; ++k) {
    int task = tid + 256 * k;         // 512 tasks
    int h = task >> 6, p = task & 63;
    __bf16* q = &qs[p * 512 + h * 64];
    float mx = -3.4e38f;
    for (int d = 0; d < 64; ++d) mx = fmaxf(mx, (float)q[d]);
    float s = 0.0f;
    for (int d = 0; d < 64; ++d) s += __expf((float)q[d] - mx);
    float inv = 1.0f / s;
    for (int d = 0; d < 64; ++d)
      q[d] = (__bf16)(__expf((float)q[d] - mx) * inv);
  }
  __syncthreads();

  // Phase B: per head, out_h[e,p] = sum_d ctx[d,e] * qs[d,p].
  int bh_base = b * NHEAD;
  for (int h = 0; h < NHEAD; ++h) {
    const float* cp = ctx + (size_t)(bh_base + h) * 4096;
    for (int idx = tid; idx < 4096; idx += 256) {
      int d = idx >> 6, e = idx & 63;
      ctxT[e * 64 + d] = (__bf16)cp[idx];    // transpose: A = ctx^T
    }
    __syncthreads();

    int e_t = wave & 3;
    int p_t0 = (wave >> 2) * 2;
    f32x8 acc[2];
    acc[0] = zero8(); acc[1] = zero8();
#pragma unroll
    for (int d0 = 0; d0 < 64; d0 += 32) {
      bf16x16 a = load_frag_a(&ctxT[e_t * 16 * 64 + d0], 64);
#pragma unroll
      for (int j = 0; j < 2; ++j) {
        bf16x16 bb = load_frag_b(&qs[(p_t0 + j) * 16 * 512 + h * 64 + d0], 512);
        acc[j] = wmma_bf16(a, bb, acc[j]);
      }
    }
    int half = lane >> 4, nn = lane & 15;
#pragma unroll
    for (int j = 0; j < 2; ++j)
#pragma unroll
      for (int r = 0; r < 8; ++r) {
        int e = e_t * 16 + r + 8 * half;
        int p = (p_t0 + j) * 16 + nn;
        outf[p * 512 + h * 64 + e] = (__bf16)acc[j][r];
      }
    __syncthreads();
  }

  // Phase C: y[c,p] = bo[c] + sum_g Wo[c,g] * outf[g,p].
  int c_t = wave;                      // 8 waves x 16 rows = 128 output channels
  f32x8 yacc[4];
#pragma unroll
  for (int i = 0; i < 4; ++i) yacc[i] = zero8();

  for (int ks = 0; ks < 16; ++ks) {
#pragma unroll
    for (int i = 0; i < 16; ++i) {     // stage Wo tile 128x32
      int idx = tid + 256 * i;
      int c = idx >> 5, kk = idx & 31;
      woA[c * 32 + kk] = (__bf16)Wo[c * 512 + ks * 32 + kk];
    }
    __syncthreads();

    bf16x16 a = load_frag_a(&woA[c_t * 16 * 32], 32);
#pragma unroll
    for (int pt = 0; pt < 4; ++pt) {
      bf16x16 bb = load_frag_b(&outf[pt * 16 * 512 + ks * 32], 512);
      yacc[pt] = wmma_bf16(a, bb, yacc[pt]);
    }
    __syncthreads();
  }

  int half = lane >> 4, nn = lane & 15;
#pragma unroll
  for (int pt = 0; pt < 4; ++pt)
#pragma unroll
    for (int r = 0; r < 8; ++r) {
      int c = c_t * 16 + r + 8 * half;
      y[(size_t)(b * CIN + c) * NTOK + p0 + pt * 16 + nn] = yacc[pt][r] + bo[c];
    }
}

// ---------------------------------------------------------------------------
extern "C" void kernel_launch(void* const* d_in, const int* in_sizes, int n_in,
                              void* d_out, int out_size, void* d_ws, size_t ws_size,
                              hipStream_t stream)
{
  const float* x  = (const float*)d_in[0];
  const float* Wq = (const float*)d_in[1];
  const float* bq = (const float*)d_in[2];
  const float* Wk = (const float*)d_in[3];
  const float* bk = (const float*)d_in[4];
  const float* Wv = (const float*)d_in[5];
  const float* bv = (const float*)d_in[6];
  const float* Wo = (const float*)d_in[7];
  const float* bo = (const float*)d_in[8];
  float* y = (float*)d_out;

  const size_t NQ = (size_t)NB * PROJROWS * NTOK;      // 33.5M elems
  __bf16* Qb = (__bf16*)d_ws;
  __bf16* Kb = Qb + NQ;
  __bf16* Vb = Kb + NQ;
  float* ctxBuf = (float*)(Vb + NQ);                   // 4*8*64*64 f32

  // 1) QKV projections: 4 batches x 3 proj x 8 row-blocks x 128 col-blocks
  qkv_proj_kernel<<<NB * 3 * 8 * 128, 256, 0, stream>>>(
      x, Wq, bq, Wk, bk, Wv, bv, Qb, Kb, Vb);

  // 2) softmax(K) over tokens, in place: 4*512 rows
  ksoftmax_kernel<<<NB * PROJROWS, 256, 0, stream>>>(Kb);

  // 3) ctx = softmax(K) @ V^T, split-K atomics
  const int CTX_ELEMS = NB * NHEAD * DK * DK;          // 131072
  zero_f32_kernel<<<(CTX_ELEMS + 255) / 256, 256, 0, stream>>>(ctxBuf, CTX_ELEMS);
  ctx_kernel<<<NB * NHEAD * 64, 256, 0, stream>>>(Kb, Vb, ctxBuf);

  // 4) fused q-softmax + per-head ctx^T@q + Wo projection
  attn_out_kernel<<<NB * (NTOK / 64), 256, 0, stream>>>(Qb, ctxBuf, Wo, bo, y);
}
